// BertSelfAttention_55800215110214
// MI455X (gfx1250) — matrix operand order repaired and checked by reading, not compiled, not run on hardware
//
#include <hip/hip_runtime.h>

typedef __attribute__((ext_vector_type(16))) _Float16 v16h;
typedef __attribute__((ext_vector_type(8)))  _Float16 v8h;
typedef __attribute__((ext_vector_type(4)))  _Float16 v4h;
typedef __attribute__((ext_vector_type(8)))  float    v8f;

#define GG  4
#define BB  4
#define LLq 512
#define CC  768
#define NHH 12
#define DHH 64

__device__ __forceinline__ v8f wmma16(v16h a, v16h b, v8f c) {
  // D = A(16x32 f16) * B(32x16 f16) + C(16x16 f32)
  return __builtin_amdgcn_wmma_f32_16x16x32_f16(false, a, false, b, (short)0, c,
                                                false, false);
}
__device__ __forceinline__ v16h cat8(v8h a, v8h b) {
  return __builtin_shufflevector(a, b, 0,1,2,3,4,5,6,7,8,9,10,11,12,13,14,15);
}
__device__ __forceinline__ float redmax16(float x) {
  x = fmaxf(x, __shfl_xor(x, 1, 32));
  x = fmaxf(x, __shfl_xor(x, 2, 32));
  x = fmaxf(x, __shfl_xor(x, 4, 32));
  x = fmaxf(x, __shfl_xor(x, 8, 32));
  return x;
}
__device__ __forceinline__ float redsum16(float x) {
  x += __shfl_xor(x, 1, 32);
  x += __shfl_xor(x, 2, 32);
  x += __shfl_xor(x, 4, 32);
  x += __shfl_xor(x, 8, 32);
  return x;
}

// ---------------- Kernel 1: fp32 -> f16 hidden states ----------------
__global__ void __launch_bounds__(256) cvt_hs_kernel(const float* __restrict__ x,
                                                     _Float16* __restrict__ y,
                                                     int n4) {
  int i = blockIdx.x * blockDim.x + threadIdx.x;
  if (i >= n4) return;
  float4 v = ((const float4*)x)[i];
  v4h o;
  o[0] = (_Float16)v.x; o[1] = (_Float16)v.y;
  o[2] = (_Float16)v.z; o[3] = (_Float16)v.w;
  ((v4h*)y)[i] = o;
}

// ---------------- Kernel 2: weight transpose+convert: wT[which][g][d][c] ----
__global__ void __launch_bounds__(256) cvt_w_kernel(const float* __restrict__ qw,
                                                    const float* __restrict__ kw,
                                                    const float* __restrict__ vw,
                                                    _Float16* __restrict__ wT) {
  __shared__ float tile[32][33];
  int blk = blockIdx.x;
  const int tc = blk % 24; blk /= 24;          // d-tile
  const int tr = blk % 24; blk /= 24;          // c-tile
  const int g  = blk & 3;  const int which = blk >> 2;
  const float* W = (which == 0 ? qw : which == 1 ? kw : vw) + (size_t)g * CC * CC;
  const int j = threadIdx.x & 31, i0 = threadIdx.x >> 5;
#pragma unroll
  for (int p = 0; p < 4; ++p) {
    int i = i0 + p * 8;
    tile[i][j] = W[(size_t)(tr * 32 + i) * CC + tc * 32 + j];  // tile[c][d]
  }
  __syncthreads();
  _Float16* o = wT + (size_t)((which * GG + g) * CC) * CC;
#pragma unroll
  for (int p = 0; p < 4; ++p) {
    int i = i0 + p * 8;
    o[(size_t)(tc * 32 + i) * CC + tr * 32 + j] = (_Float16)tile[j][i];
  }
}

// ---------------- Kernel 3: QKV projection GEMM (WMMA) ----------------
// One wave computes a 32(M) x 64(N) strip: 8 WMMAs per 32-deep k-chunk,
// B fragments reused across both M subtiles. Manual ping-pong (unroll-by-2)
// double buffering: no buffer-rotation copies, loads of one chunk in flight
// while the other chunk's WMMA burst executes.
__device__ __forceinline__ void load_chunk(const _Float16* __restrict__ ap,
                                           const _Float16* __restrict__ bp,
                                           int lo, int hi, int ko,
                                           v16h& a0, v16h& a1, v16h b[4]) {
  a0 = cat8(*(const v8h*)(ap + (size_t)lo * CC + ko + hi * 16),
            *(const v8h*)(ap + (size_t)lo * CC + ko + 8 + hi * 16));
  a1 = cat8(*(const v8h*)(ap + (size_t)(16 + lo) * CC + ko + hi * 16),
            *(const v8h*)(ap + (size_t)(16 + lo) * CC + ko + 8 + hi * 16));
#pragma unroll
  for (int t = 0; t < 4; ++t)
    b[t] = *(const v16h*)(bp + (size_t)(t * 16 + lo) * CC + ko + hi * 16);
}

__device__ __forceinline__ void burst8(v8f acc[2][4], const v16h& a0,
                                       const v16h& a1, const v16h b[4]) {
#pragma unroll
  for (int t = 0; t < 4; ++t) acc[0][t] = wmma16(a0, b[t], acc[0][t]);
#pragma unroll
  for (int t = 0; t < 4; ++t) acc[1][t] = wmma16(a1, b[t], acc[1][t]);
}

__global__ void __launch_bounds__(256) qkv_kernel(
    const _Float16* __restrict__ hsH, const _Float16* __restrict__ wT,
    const float* __restrict__ qb, const float* __restrict__ kb,
    const float* __restrict__ vb,
    _Float16* __restrict__ qH, _Float16* __restrict__ kH,
    _Float16* __restrict__ vtH) {
  const int lane = threadIdx.x & 31;
  const int lo = lane & 15, hi = lane >> 4;
  int idx = blockIdx.x * 8 + (threadIdx.x >> 5);
  const int ns = idx % 12; idx /= 12;       // N strip of 64 (== head index)
  const int mt = idx & 63; idx >>= 6;       // M tile of 32 (of 2048 tokens)
  const int which = idx % 3; const int g = idx / 3;

  const int row0 = mt * 32;
  const _Float16* ap = hsH + ((size_t)(g * 2048 + row0)) * CC;
  const _Float16* bp = wT + ((size_t)((which * GG + g) * CC + ns * 64)) * CC;
  const float* biasp = (which == 0 ? qb : which == 1 ? kb : vb) + g * CC + ns * 64;

  v8f acc[2][4];
  const v8f zf = {0.f, 0.f, 0.f, 0.f, 0.f, 0.f, 0.f, 0.f};
#pragma unroll
  for (int mi = 0; mi < 2; ++mi)
#pragma unroll
    for (int t = 0; t < 4; ++t) acc[mi][t] = zf;

  v16h xa0, xa1, xb[4];   // ping
  v16h ya0, ya1, yb[4];   // pong
  load_chunk(ap, bp, lo, hi, 0, xa0, xa1, xb);

  for (int kc = 0; kc < 24; kc += 2) {
    load_chunk(ap, bp, lo, hi, (kc + 1) * 32, ya0, ya1, yb);
    burst8(acc, xa0, xa1, xb);
    const int k2 = (kc == 22) ? 0 : (kc + 2) * 32;  // wrap keeps it in-bounds
    load_chunk(ap, bp, lo, hi, k2, xa0, xa1, xb);
    burst8(acc, ya0, ya1, yb);
  }

  const int b_ = row0 >> 9, l0 = row0 & 511;
  const size_t head = (size_t)((g * BB + b_) * NHH + ns);
  if (which < 2) {
    _Float16* dst = (which == 0 ? qH : kH) + (head * LLq + l0) * DHH;
#pragma unroll
    for (int t = 0; t < 4; ++t) {
      const float bias = biasp[t * 16 + lo];
      const int dh = t * 16 + lo;
#pragma unroll
      for (int mi = 0; mi < 2; ++mi)
#pragma unroll
        for (int v = 0; v < 8; ++v)
          dst[(size_t)(mi * 16 + v + hi * 8) * DHH + dh] =
              (_Float16)(acc[mi][t][v] + bias);
    }
  } else {
    // V stored transposed: vtH[head][dh][l] -> contiguous 8-half packed stores
    _Float16* dst = vtH + head * DHH * LLq + l0;
#pragma unroll
    for (int t = 0; t < 4; ++t) {
      const float bias = biasp[t * 16 + lo];
      const int dh = t * 16 + lo;
#pragma unroll
      for (int mi = 0; mi < 2; ++mi) {
        v8h pack;
#pragma unroll
        for (int v = 0; v < 8; ++v) pack[v] = (_Float16)(acc[mi][t][v] + bias);
        *(v8h*)(dst + (size_t)dh * LLq + mi * 16 + hi * 8) = pack;
      }
    }
  }
}

// ---------------- Kernel 4: flash attention (WMMA + online softmax) --------
// One wave per (g,b,h, 16-query tile). 32-key chunks; next chunk's K loads
// issue right after the score WMMAs and overlap the softmax section.
__global__ void __launch_bounds__(256) attn_kernel(
    const _Float16* __restrict__ qH, const _Float16* __restrict__ kH,
    const _Float16* __restrict__ vtH, const float* __restrict__ mask,
    float* __restrict__ out) {
  // per-wave probs tile, 16 rows x 32 cols f16, row stride 40 halves (80B):
  // 16B-aligned fragment reads, conflict-free (20*lo mod 64 distinct).
  __shared__ __align__(64) _Float16 pbuf_all[8][16 * 40];
  const int wave = threadIdx.x >> 5;
  const int lane = threadIdx.x & 31;
  const int lo = lane & 15, hi = lane >> 4;
  int idx = blockIdx.x * 8 + wave;
  const int qt = idx & 31; idx >>= 5;
  const int h = idx % NHH; idx /= NHH;
  const int b = idx & 3; const int g = idx >> 2;

  const size_t head = (size_t)((g * BB + b) * NHH + h);
  const _Float16* qp = qH + (head * LLq + qt * 16) * DHH;
  const _Float16* kp = kH + head * LLq * DHH;
  const _Float16* vp = vtH + head * DHH * LLq;
  const float* mp = mask + (size_t)(g * BB + b) * LLq;
  _Float16* pb = &pbuf_all[wave][0];

  // Q A-fragments (reused over all key chunks), K-dim = dh (2 steps of 32)
  v16h aq0 = cat8(*(const v8h*)(qp + lo * DHH + hi * 16),
                  *(const v8h*)(qp + lo * DHH + 8 + hi * 16));
  v16h aq1 = cat8(*(const v8h*)(qp + lo * DHH + 32 + hi * 16),
                  *(const v8h*)(qp + lo * DHH + 40 + hi * 16));

  const v8f zf = {0.f, 0.f, 0.f, 0.f, 0.f, 0.f, 0.f, 0.f};
  v8f ctx[4];
#pragma unroll
  for (int t = 0; t < 4; ++t) ctx[t] = zf;
  float rowmax[8], rowsum[8], fscale[8];
#pragma unroll
  for (int v = 0; v < 8; ++v) { rowmax[v] = -3.0e38f; rowsum[v] = 0.f; }

  // prologue: K fragments for chunk 0
  v16h bk[4];
  bk[0] = *(const v16h*)(kp + (size_t)lo * DHH + hi * 16);
  bk[1] = *(const v16h*)(kp + (size_t)lo * DHH + 32 + hi * 16);
  bk[2] = *(const v16h*)(kp + (size_t)(16 + lo) * DHH + hi * 16);
  bk[3] = *(const v16h*)(kp + (size_t)(16 + lo) * DHH + 32 + hi * 16);

  for (int c = 0; c < 16; ++c) {       // 16 chunks x 32 keys = 512
    const int kb0 = c * 32;
    v8f s0 = zf, s1 = zf;
    s0 = wmma16(aq0, bk[0], s0); s0 = wmma16(aq1, bk[1], s0);
    s1 = wmma16(aq0, bk[2], s1); s1 = wmma16(aq1, bk[3], s1);

    // issue next chunk's K loads; they overlap the softmax section below
    const int cn = (c == 15) ? 0 : (c + 1) * 32;
    bk[0] = *(const v16h*)(kp + (size_t)(cn + lo) * DHH + hi * 16);
    bk[1] = *(const v16h*)(kp + (size_t)(cn + lo) * DHH + 32 + hi * 16);
    bk[2] = *(const v16h*)(kp + (size_t)(cn + 16 + lo) * DHH + hi * 16);
    bk[3] = *(const v16h*)(kp + (size_t)(cn + 16 + lo) * DHH + 32 + hi * 16);

    const float m0 = mp[kb0 + lo], m1 = mp[kb0 + 16 + lo];
#pragma unroll
    for (int v = 0; v < 8; ++v) {
      // lane holds row (v + hi*8), column lo / lo+16 of this chunk
      float sa = s0[v] * 0.125f + m0;
      float sb = s1[v] * 0.125f + m1;
      float cm = redmax16(fmaxf(sa, sb));        // max over 16 cols (same hi-half)
      float nm = fmaxf(rowmax[v], cm);
      float f = __expf(rowmax[v] - nm);
      rowmax[v] = nm;
      float p0 = __expf(sa - nm), p1 = __expf(sb - nm);
      rowsum[v] = rowsum[v] * f + redsum16(p0 + p1);
      fscale[v] = f;
      pb[(v + hi * 8) * 40 + lo] = (_Float16)p0;
      pb[(v + hi * 8) * 40 + 16 + lo] = (_Float16)p1;
    }
#pragma unroll
    for (int t = 0; t < 4; ++t)
#pragma unroll
      for (int v = 0; v < 8; ++v) ctx[t][v] *= fscale[v];

    // LDS stores (other lanes' data) must land before fragment loads
    asm volatile("s_wait_dscnt 0" ::: "memory");

    // P A-fragment: M=query row lo, K=32 keys of this chunk
    v16h apf = cat8(*(const v8h*)(pb + lo * 40 + hi * 16),
                    *(const v8h*)(pb + lo * 40 + 8 + hi * 16));
#pragma unroll
    for (int t = 0; t < 4; ++t) {
      // V B-fragment: lane = dh column, contiguous keys (vtH is [dh][l])
      v16h bv = *(const v16h*)(vp + (size_t)(t * 16 + lo) * LLq + kb0 + hi * 16);
      ctx[t] = wmma16(apf, bv, ctx[t]);
    }
  }

  // normalize + store: out[g][b][l][h*64 + dh], fp32
  const size_t outbase = ((size_t)((g * BB + b) * LLq + qt * 16)) * CC + h * DHH;
#pragma unroll
  for (int v = 0; v < 8; ++v) {
    const float inv = 1.0f / rowsum[v];
    float* op = out + outbase + (size_t)(v + hi * 8) * CC;
#pragma unroll
    for (int t = 0; t < 4; ++t) op[t * 16 + lo] = ctx[t][v] * inv;
  }
}

// ---------------- Host launcher ----------------
extern "C" void kernel_launch(void* const* d_in, const int* in_sizes, int n_in,
                              void* d_out, int out_size, void* d_ws, size_t ws_size,
                              hipStream_t stream) {
  const float* hs    = (const float*)d_in[0];
  const float* mask  = (const float*)d_in[1];
  const float* qw    = (const float*)d_in[2];
  const float* qbias = (const float*)d_in[3];
  const float* kw    = (const float*)d_in[4];
  const float* kbias = (const float*)d_in[5];
  const float* vw    = (const float*)d_in[6];
  const float* vbias = (const float*)d_in[7];
  float* out = (float*)d_out;

  const size_t nHS = (size_t)GG * BB * LLq * CC;  // 6,291,456
  const size_t nW  = (size_t)3 * GG * CC * CC;    // 7,077,888
  _Float16* hsH = (_Float16*)d_ws;
  _Float16* wT  = hsH + nHS;
  _Float16* qH  = wT + nW;
  _Float16* kH  = qH + nHS;
  _Float16* vtH = kH + nHS;   // total ~64.5 MB of the workspace

  cvt_hs_kernel<<<(int)(nHS / 4 / 256), 256, 0, stream>>>(hs, hsH, (int)(nHS / 4));
  cvt_w_kernel<<<3 * 4 * 24 * 24, 256, 0, stream>>>(qw, kw, vw, wT);
  // waves: 4 groups * 3 mats * 64 Mtiles(32 rows) * 12 Nstrips = 9216 -> 1152 blocks
  qkv_kernel<<<1152, 256, 0, stream>>>(hsH, wT, qbias, kbias, vbias, qH, kH, vtH);
  // waves: 4*4*12 heads * 32 qtiles = 6144 -> 768 blocks
  attn_kernel<<<768, 256, 0, stream>>>(qH, kH, vtH, mask, out);
}
